// GPTAttention_73864847557017
// MI455X (gfx1250) — matrix operand order repaired and checked by reading, not compiled, not run on hardware
//
#include <hip/hip_runtime.h>
#include <hip/hip_bf16.h>

typedef __attribute__((ext_vector_type(16))) _Float16 v16h;
typedef __attribute__((ext_vector_type(8)))  _Float16 v8h;
typedef __attribute__((ext_vector_type(8)))  float    v8f;
typedef __attribute__((ext_vector_type(4)))  float    v4f;

// Problem constants
#define BB   2
#define SS   2048
#define DD   2048
#define HH   32
#define DHH  64
#define MM   (BB * SS)   // 4096 rows

// ---------- helpers ----------

// A fragment from f16 source: elements 0..7 = k+0..7, elements 8..15 = k+16..23
__device__ inline v16h loadA_h(const _Float16* __restrict__ p) {
    v16h r;
    v8h lo = *(const v8h*)(p);
    v8h hi = *(const v8h*)(p + 16);
#pragma unroll
    for (int i = 0; i < 8; i++) { r[i] = lo[i]; r[i + 8] = hi[i]; }
    return r;
}

// 16 contiguous halfs (B fragment: k contiguous per lane)
__device__ inline v16h load16_h(const _Float16* p) {
    v16h r;
    v8h lo = *(const v8h*)(p);
    v8h hi = *(const v8h*)(p + 8);
#pragma unroll
    for (int i = 0; i < 8; i++) { r[i] = lo[i]; r[i + 8] = hi[i]; }
    return r;
}

__device__ inline float hmax16(float v) {
#pragma unroll
    for (int d = 1; d < 16; d <<= 1) v = fmaxf(v, __shfl_xor(v, d, 32));
    return v;
}
__device__ inline float hsum16(float v) {
#pragma unroll
    for (int d = 1; d < 16; d <<= 1) v += __shfl_xor(v, d, 32);
    return v;
}

// CDNA5 async global->LDS copy (16B per lane), tracked by ASYNCcnt.
__device__ inline void async_copy16(_Float16* ldsPtr, const _Float16* gPtr) {
    unsigned int loff = (unsigned int)(size_t)ldsPtr;       // LDS aperture: low 32 bits
    unsigned long long ga = (unsigned long long)(size_t)gPtr;
    asm volatile("global_load_async_to_lds_b128 %0, %1, off"
                 :: "v"(loff), "v"(ga)
                 : "memory");
}
__device__ inline void wait_async0() {
    asm volatile("s_wait_asynccnt 0x0" ::: "memory");
}

// ---------- Kernel 0: one-time f32 -> f16 conversion (bandwidth-bound) ----------
__global__ __launch_bounds__(256)
void cvt_f32_to_f16(const float* __restrict__ src, _Float16* __restrict__ dst, int n8) {
    int i = blockIdx.x * blockDim.x + threadIdx.x;
    if (i >= n8) return;
    const float* p = src + (size_t)i * 8;
    v4f x = *(const v4f*)(p);
    v4f y = *(const v4f*)(p + 4);
    v8h o;
#pragma unroll
    for (int j = 0; j < 4; j++) { o[j] = (_Float16)x[j]; o[j + 4] = (_Float16)y[j]; }
    *(v8h*)(dst + (size_t)i * 8) = o;
}

// ---------- Kernel 1: QKV projection  C = Xh @ W^T + b, all-f16 operands ----------
// Wave computes a 32(M) x 64(N) tile: 8 WMMAs per K-step, B-frags reused x2.
// mode 0: Q -> [b,h,s,dh] scaled by DH^-0.5
// mode 1: K -> [b,h,s,dh]
// mode 2: V -> [b,h,dh,s]   (transposed so PV B-fragments are contiguous)
__global__ __launch_bounds__(256)
void proj_qkv_kernel(const _Float16* __restrict__ A, const _Float16* __restrict__ W,
                     const float* __restrict__ bias, _Float16* __restrict__ out, int mode) {
    const int wid  = (blockIdx.x * blockDim.x + threadIdx.x) >> 5;
    const int lane = threadIdx.x & 31;
    const int hi   = lane >> 4;
    const int l15  = lane & 15;

    const int nStrips = DD / 64;            // 32
    const int mTiles  = MM / 32;            // 128
    if (wid >= mTiles * nStrips) return;
    const int m0 = (wid / nStrips) * 32;
    const int n0 = (wid % nStrips) * 64;

    v8f acc0[4] = {};
    v8f acc1[4] = {};

    const _Float16* ap0 = A + (size_t)(m0 + l15) * DD;
    const _Float16* ap1 = A + (size_t)(m0 + 16 + l15) * DD;

    for (int k0 = 0; k0 < DD; k0 += 32) {
        v16h a0 = loadA_h(ap0 + k0 + hi * 8);
        v16h a1 = loadA_h(ap1 + k0 + hi * 8);
        if (k0 + 32 < DD) {
            __builtin_prefetch(ap0 + k0 + 32 + hi * 8, 0, 1);
            __builtin_prefetch(ap1 + k0 + 32 + hi * 8, 0, 1);
        }
#pragma unroll
        for (int t = 0; t < 4; t++) {
            const int ncol = n0 + t * 16 + l15;
            v16h bfrag = loadA_h(W + (size_t)ncol * DD + k0 + hi * 8);
            acc0[t] = __builtin_amdgcn_wmma_f32_16x16x32_f16(
                false, a0, false, bfrag, (short)0, acc0[t], false, false);
            acc1[t] = __builtin_amdgcn_wmma_f32_16x16x32_f16(
                false, a1, false, bfrag, (short)0, acc1[t], false, false);
        }
    }

#pragma unroll
    for (int t = 0; t < 4; t++) {
        const int ncol = n0 + t * 16 + l15;
        const float bv = bias[ncol];
        const int hh = ncol >> 6;     // head
        const int dh = ncol & 63;
#pragma unroll
        for (int half = 0; half < 2; half++) {
            const v8f* accp = half ? acc1 : acc0;
#pragma unroll
            for (int r = 0; r < 8; r++) {
                const int mrow = m0 + half * 16 + r + hi * 8;
                const int bb = mrow >> 11;         // /S
                const int ss = mrow & (SS - 1);
                float val = accp[t][r] + bv;
                if (mode == 0) val *= 0.125f;      // DH^-0.5
                size_t idx;
                if (mode == 2)
                    idx = (((size_t)(bb * HH + hh)) * DHH + dh) * SS + ss;
                else
                    idx = (((size_t)(bb * HH + hh)) * SS + ss) * DHH + dh;
                out[idx] = (_Float16)val;
            }
        }
    }
}

// ---------- Kernel 2: block-cooperative causal flash attention ----------
// Block = (b,h) x 128 q-rows (8 waves, 16 rows each). All waves share the same
// K/V stream: 32-key K (32x64 f16) and V (64x32 f16) tiles are staged into LDS
// once per block via async global->LDS copies, double-buffered.
__global__ __launch_bounds__(256)
void attn_kernel(const _Float16* __restrict__ Qh, const _Float16* __restrict__ Kh,
                 const _Float16* __restrict__ Vt, _Float16* __restrict__ Oh) {
    __shared__ _Float16 kbuf[2][32 * DHH];   // 2 x 4 KB
    __shared__ _Float16 vbuf[2][DHH * 32];   // 2 x 4 KB
    __shared__ _Float16 plds[8][16][32];     // per-wave P re-layout buffer (8 KB)

    const int tid  = threadIdx.x;
    const int lane = tid & 31;
    const int hi   = lane >> 4;
    const int l15  = lane & 15;
    const int wslot = tid >> 5;

    const int qBlocks = SS / 128;            // 16
    const int bh    = blockIdx.x / qBlocks;  // b*H + h
    const int qBase = (blockIdx.x % qBlocks) * 128;
    const int b  = bh / HH;
    const int h  = bh % HH;
    const int q0 = qBase + wslot * 16;       // this wave's 16 q-rows

    const _Float16* Kbase = Kh + (size_t)bh * SS * DHH;          // [key][dh]
    const _Float16* Vbase = Vt + (size_t)bh * DHH * SS;          // [dh][s]

    // staging assignments (per thread, 16B each for K and V)
    const int krow = tid >> 3, kcol = (tid & 7) * 8;             // 32 x 64
    const int vrow = tid >> 2, vcol = (tid & 3) * 8;             // 64 x 32

    // Q fragments: Q[16 x 64] = 2 A-frags (K-steps of 32 over dh)
    v16h qf[2];
    const _Float16* qbase = Qh + ((size_t)bh * SS + q0) * DHH;
#pragma unroll
    for (int kk = 0; kk < 2; kk++)
        qf[kk] = loadA_h(qbase + (size_t)l15 * DHH + kk * 32 + hi * 8);

    v8f o[4] = {};
    float mrow[8], lrow[8];
#pragma unroll
    for (int r = 0; r < 8; r++) { mrow[r] = -3.0e38f; lrow[r] = 0.0f; }

    const int nblocks = qBase / 32 + 4;      // covers keys [0, qBase+128)

    // prime buffer 0 with key block 0
    async_copy16(&kbuf[0][krow * DHH + kcol], Kbase + (size_t)krow * DHH + kcol);
    async_copy16(&vbuf[0][vrow * 32 + vcol], Vbase + (size_t)vrow * SS + vcol);

    for (int jb = 0; jb < nblocks; jb++) {
        const int cur = jb & 1;
        const int j0  = jb * 32;

        wait_async0();               // our stage of buf[cur] is complete
        __syncthreads();             // everyone's stage of buf[cur] visible

        if (jb + 1 < nblocks) {      // stream next tile into the other buffer
            const int jn = (jb + 1) * 32;
            async_copy16(&kbuf[cur ^ 1][krow * DHH + kcol],
                         Kbase + ((size_t)(jn + krow)) * DHH + kcol);
            async_copy16(&vbuf[cur ^ 1][vrow * 32 + vcol],
                         Vbase + (size_t)vrow * SS + jn + vcol);
        }

        if (j0 <= q0 + 15) {         // wave-uniform causal relevance
            // scores: two 16x16 tiles covering keys [j0, j0+32)
            v8f st[2] = {};
#pragma unroll
            for (int t = 0; t < 2; t++) {
#pragma unroll
                for (int kk = 0; kk < 2; kk++) {
                    v16h kfrag = loadA_h(&kbuf[cur][(t * 16 + l15) * DHH + kk * 32 + hi * 8]);
                    st[t] = __builtin_amdgcn_wmma_f32_16x16x32_f16(
                        false, qf[kk], false, kfrag, (short)0, st[t], false, false);
                }
            }

            // causal mask (only needed on diagonal blocks)
            if (j0 + 31 > q0) {
#pragma unroll
                for (int t = 0; t < 2; t++) {
                    const int key = j0 + t * 16 + l15;
#pragma unroll
                    for (int r = 0; r < 8; r++) {
                        const int qrow = q0 + r + hi * 8;
                        if (key > qrow) st[t][r] = -1.0e9f;
                    }
                }
            }

            // online softmax (rows live across 16 lanes of each half-wave)
#pragma unroll
            for (int r = 0; r < 8; r++) {
                float mx = hmax16(fmaxf(st[0][r], st[1][r]));
                float mnew = fmaxf(mrow[r], mx);
                float alpha = __expf(mrow[r] - mnew);
                mrow[r] = mnew;
                float p0 = __expf(st[0][r] - mnew);
                float p1 = __expf(st[1][r] - mnew);
                st[0][r] = p0; st[1][r] = p1;
                lrow[r] = lrow[r] * alpha + hsum16(p0 + p1);
#pragma unroll
                for (int t = 0; t < 4; t++) o[t][r] *= alpha;
            }

            // re-layout P (C-layout) -> A-fragment via LDS
#pragma unroll
            for (int t = 0; t < 2; t++)
#pragma unroll
                for (int r = 0; r < 8; r++)
                    plds[wslot][r + hi * 8][t * 16 + l15] = (_Float16)st[t][r];

            v16h pf = loadA_h(&plds[wslot][l15][hi * 8]);

            // O += P @ V
#pragma unroll
            for (int d = 0; d < 4; d++) {
                v16h vfrag = loadA_h(&vbuf[cur][(d * 16 + l15) * 32 + hi * 8]);
                o[d] = __builtin_amdgcn_wmma_f32_16x16x32_f16(
                    false, pf, false, vfrag, (short)0, o[d], false, false);
            }
        }

        __syncthreads();             // all waves done reading buf[cur]
    }

    // normalize + store f16 attention output in [b, s, h*64+dh]
    float inv[8];
#pragma unroll
    for (int r = 0; r < 8; r++) inv[r] = 1.0f / lrow[r];
#pragma unroll
    for (int d = 0; d < 4; d++) {
        const int dcol = h * 64 + d * 16 + l15;
#pragma unroll
        for (int r = 0; r < 8; r++) {
            const int srow = q0 + r + hi * 8;
            Oh[((size_t)b * SS + srow) * DD + dcol] = (_Float16)(o[d][r] * inv[r]);
        }
    }
}

// ---------- Kernel 3: output projection  out = Oh @ o_w^T + o_b (f32 out) ----------
__global__ __launch_bounds__(256)
void out_proj_kernel(const _Float16* __restrict__ A, const _Float16* __restrict__ W,
                     const float* __restrict__ bias, float* __restrict__ out) {
    const int wid  = (blockIdx.x * blockDim.x + threadIdx.x) >> 5;
    const int lane = threadIdx.x & 31;
    const int hi   = lane >> 4;
    const int l15  = lane & 15;

    const int nStrips = DD / 64;
    const int mTiles  = MM / 32;
    if (wid >= mTiles * nStrips) return;
    const int m0 = (wid / nStrips) * 32;
    const int n0 = (wid % nStrips) * 64;

    v8f acc0[4] = {};
    v8f acc1[4] = {};

    const _Float16* ap0 = A + (size_t)(m0 + l15) * DD;
    const _Float16* ap1 = A + (size_t)(m0 + 16 + l15) * DD;

    for (int k0 = 0; k0 < DD; k0 += 32) {
        v16h a0 = loadA_h(ap0 + k0 + hi * 8);
        v16h a1 = loadA_h(ap1 + k0 + hi * 8);
        if (k0 + 32 < DD) {
            __builtin_prefetch(ap0 + k0 + 32 + hi * 8, 0, 1);
            __builtin_prefetch(ap1 + k0 + 32 + hi * 8, 0, 1);
        }
#pragma unroll
        for (int t = 0; t < 4; t++) {
            const int ncol = n0 + t * 16 + l15;
            v16h bfrag = loadA_h(W + (size_t)ncol * DD + k0 + hi * 8);
            acc0[t] = __builtin_amdgcn_wmma_f32_16x16x32_f16(
                false, a0, false, bfrag, (short)0, acc0[t], false, false);
            acc1[t] = __builtin_amdgcn_wmma_f32_16x16x32_f16(
                false, a1, false, bfrag, (short)0, acc1[t], false, false);
        }
    }

#pragma unroll
    for (int t = 0; t < 4; t++) {
        const int ncol = n0 + t * 16 + l15;
        const float bv = bias[ncol];
#pragma unroll
        for (int r = 0; r < 8; r++) {
            const int mrow0 = m0 + r + hi * 8;
            const int mrow1 = m0 + 16 + r + hi * 8;
            out[(size_t)mrow0 * DD + ncol] = acc0[t][r] + bv;
            out[(size_t)mrow1 * DD + ncol] = acc1[t][r] + bv;
        }
    }
}

// ---------- launch ----------
extern "C" void kernel_launch(void* const* d_in, const int* in_sizes, int n_in,
                              void* d_out, int out_size, void* d_ws, size_t ws_size,
                              hipStream_t stream) {
    const float* hidden = (const float*)d_in[0];
    // d_in[1] = additive causal mask: applied analytically, never read
    const float* q_w = (const float*)d_in[2];
    const float* q_b = (const float*)d_in[3];
    const float* k_w = (const float*)d_in[4];
    const float* k_b = (const float*)d_in[5];
    const float* v_w = (const float*)d_in[6];
    const float* v_b = (const float*)d_in[7];
    const float* o_w = (const float*)d_in[8];
    const float* o_b = (const float*)d_in[9];

    // f16 workspace layout
    const size_t hidElems = (size_t)MM * DD;           // 8,388,608
    const size_t wElems   = (size_t)DD * DD;           // 4,194,304
    _Float16* Xh = (_Float16*)d_ws;
    _Float16* Wq = Xh + hidElems;
    _Float16* Wk = Wq + wElems;
    _Float16* Wv = Wk + wElems;
    _Float16* Wo = Wv + wElems;
    _Float16* Qh = Wo + wElems;
    _Float16* Kh = Qh + hidElems;
    _Float16* Vt = Kh + hidElems;
    _Float16* Oh = Vt + hidElems;

    // one-time conversions (bandwidth-bound, ~8 us total at 23.3 TB/s)
    {
        int n8h = (int)(hidElems / 8);
        int n8w = (int)(wElems / 8);
        cvt_f32_to_f16<<<(n8h + 255) / 256, 256, 0, stream>>>(hidden, Xh, n8h);
        cvt_f32_to_f16<<<(n8w + 255) / 256, 256, 0, stream>>>(q_w, Wq, n8w);
        cvt_f32_to_f16<<<(n8w + 255) / 256, 256, 0, stream>>>(k_w, Wk, n8w);
        cvt_f32_to_f16<<<(n8w + 255) / 256, 256, 0, stream>>>(v_w, Wv, n8w);
        cvt_f32_to_f16<<<(n8w + 255) / 256, 256, 0, stream>>>(o_w, Wo, n8w);
    }

    // GEMM waves: 128 m-tiles * 32 n-strips = 4096 waves -> 512 blocks of 256
    const int gemmBlocks = (MM / 32) * (DD / 64) / 8;
    proj_qkv_kernel<<<gemmBlocks, 256, 0, stream>>>(Xh, Wq, q_b, Qh, 0);
    proj_qkv_kernel<<<gemmBlocks, 256, 0, stream>>>(Xh, Wk, k_b, Kh, 1);
    proj_qkv_kernel<<<gemmBlocks, 256, 0, stream>>>(Xh, Wv, v_b, Vt, 2);

    // attention: (B*H) x (S/128) = 1024 blocks of 8 cooperating waves
    const int attnBlocks = BB * HH * (SS / 128);
    attn_kernel<<<attnBlocks, 256, 0, stream>>>(Qh, Kh, Vt, Oh);

    out_proj_kernel<<<gemmBlocks, 256, 0, stream>>>(Oh, Wo, o_b, (float*)d_out);
}
